// GNN_7730941133172
// MI455X (gfx1250) — matrix operand order repaired and checked by reading, not compiled, not run on hardware
//
#include <hip/hip_runtime.h>
#include <hip/hip_bf16.h>

// ---------------------------------------------------------------------------
// 2-layer GCN for MI455X (gfx1250, wave32, WMMA).
//   L1: H0 = X(89250x500) @ W1(500x64)        -> bf16 WMMA, f32 accum
//       A1 = D^-1/2 A_hat D^-1/2 H0 ; H1 = relu(A1 + b1)
//   L2: H2 = H1(89250x64) @ W2(64x7)          -> bf16 WMMA (N padded to 16)
//       out = D^-1/2 A_hat D^-1/2 H2 + b2
//
// W matrices are pre-packed in LDS into the exact WMMA B-fragment register
// image so each lane fetches its v16bf fragment with wide ds_load_b128s.
// ---------------------------------------------------------------------------

#define FDIM 500
#define KPAD 512
#define HDIM 64
#define CDIM 7

typedef __attribute__((ext_vector_type(16))) __bf16 v16bf;
typedef __attribute__((ext_vector_type(8)))  float  v8f;

static __device__ __forceinline__ void atomic_add_f32(float* p, float v) {
  __hip_atomic_fetch_add(p, v, __ATOMIC_RELAXED, __HIP_MEMORY_SCOPE_AGENT);
}

// ---------------------------------------------------------------------------
// Degree / normalization
// ---------------------------------------------------------------------------
__global__ void init_deg_kernel(float* deg, int n) {
  int i = blockIdx.x * 256 + threadIdx.x;
  if (i < n) deg[i] = 1.0f;  // self loop
}

__global__ void deg_edges_kernel(const long long* __restrict__ dst, float* deg, int e) {
  int i = blockIdx.x * 256 + threadIdx.x;
  if (i < e) atomic_add_f32(&deg[(int)dst[i]], 1.0f);
}

__global__ void dinv_kernel(const float* __restrict__ deg, float* dinv, int n) {
  int i = blockIdx.x * 256 + threadIdx.x;
  if (i < n) {
    float d = deg[i];
    dinv[i] = d > 0.0f ? rsqrtf(d) : 0.0f;
  }
}

// ---------------------------------------------------------------------------
// GEMM1: H0[n x 64] = X[n x 500] @ W1[500 x 64]   (bf16 WMMA, f32 accum)
// Block = 256 threads = 8 waves; wave owns a 16x64 tile (4 x 16x16 accum).
// sW holds W1 pre-packed as per-lane B fragments:
//   element j of (kstep s, tile t, lane l) = W1[s*32 + (l>=16)*8 + (j&7) + (j>>3)*16][t*16 + (l&15)]
//   at sW[(((s*4 + t)*32 + l)*16 + j]
// ---------------------------------------------------------------------------
template <bool GUARD>
static __device__ __forceinline__ void gemm1_step(const float* __restrict__ xrow,
                                                  const __bf16* __restrict__ sWp,
                                                  int k0, int lane, int lane_m, int half,
                                                  v8f acc[4]) {
  // A fragment, 16-bit A 16x32 layout: elements [0..7] = K kb..kb+7,
  // [8..15] = K kb+16..kb+23, kb = k0 + 8*half.
  int kb = k0 + half * 8;
  v16bf a;
#pragma unroll
  for (int j = 0; j < 8; ++j) {
    int k1 = kb + j, k2 = kb + 16 + j;
    float f1 = (!GUARD || k1 < FDIM) ? xrow[k1] : 0.0f;
    float f2 = (!GUARD || k2 < FDIM) ? xrow[k2] : 0.0f;
    a[j]     = (__bf16)f1;
    a[8 + j] = (__bf16)f2;
  }
  int s = k0 >> 5;
#pragma unroll
  for (int t = 0; t < 4; ++t) {
    const v16bf b = *(const v16bf*)(sWp + (size_t)(((s * 4 + t) * 32 + lane) * 16));
    acc[t] = __builtin_amdgcn_wmma_f32_16x16x32_bf16(
        false, a, false, b, (short)0, acc[t], false, false);
  }
}

__global__ __launch_bounds__(256) void gemm1_kernel(const float* __restrict__ X,
                                                    const float* __restrict__ W1,
                                                    float* __restrict__ H0, int n) {
  __shared__ __bf16 sW[KPAD * HDIM];  // 64 KB, B-fragment packed
  int tid = threadIdx.x;
  for (int idx = tid; idx < KPAD * HDIM; idx += 256) {
    int k = idx >> 6, c = idx & 63;
    float v = (k < FDIM) ? W1[idx] : 0.0f;
    int s = k >> 5, jj = k & 31;
    int hk = (jj >> 3) & 1, j = (jj & 7) + ((jj >> 4) << 3);
    int t = c >> 4, lane = (c & 15) + hk * 16;
    sW[(((s * 4 + t) * 32 + lane) * 16) + j] = (__bf16)v;
  }
  __syncthreads();

  int wave = tid >> 5, lane = tid & 31;
  int lane_m = lane & 15, half = lane >> 4;
  int row0 = blockIdx.x * 128 + wave * 16;

  int arow = row0 + lane_m;
  if (arow >= n) arow = n - 1;  // clamp loads; stores are guarded
  const float* xrow = X + (size_t)arow * FDIM;

  v8f acc[4] = {v8f{}, v8f{}, v8f{}, v8f{}};

#pragma unroll 2
  for (int k0 = 0; k0 < KPAD - 32; k0 += 32) {  // k <= 479 < 500: unguarded
    __builtin_prefetch(xrow + k0 + 32, 0, 0);   // global_prefetch_b8
    gemm1_step<false>(xrow, sW, k0, lane, lane_m, half, acc);
  }
  gemm1_step<true>(xrow, sW, KPAD - 32, lane, lane_m, half, acc);  // tail, guarded

  // C/D layout: VGPR i -> M = i + 8*half, N = lane_m
#pragma unroll
  for (int t = 0; t < 4; ++t) {
#pragma unroll
    for (int i = 0; i < 8; ++i) {
      int r = row0 + half * 8 + i;
      if (r < n) H0[(size_t)r * HDIM + t * 16 + lane_m] = acc[t][i];
    }
  }
}

// ---------------------------------------------------------------------------
// Aggregation, layer 1 (64 features)
// ---------------------------------------------------------------------------
__global__ void agg1_init_kernel(const float* __restrict__ dinv, const float* __restrict__ H0,
                                 float* __restrict__ A1, int n) {
  long long idx = (long long)blockIdx.x * 256 + threadIdx.x;
  if (idx >= (long long)n * HDIM) return;
  int i = (int)(idx >> 6);
  float di = dinv[i];
  A1[idx] = di * di * H0[idx];  // self-loop message
}

__global__ void agg1_edges_kernel(const long long* __restrict__ src,
                                  const long long* __restrict__ dst,
                                  const float* __restrict__ dinv,
                                  const float* __restrict__ H0,
                                  float* __restrict__ A1, long long total) {
  long long tid = (long long)blockIdx.x * 256 + threadIdx.x;
  if (tid >= total) return;
  long long e = tid >> 6;
  int f = (int)(tid & 63);
  int s = (int)src[e], d = (int)dst[e];
  float w = dinv[s] * dinv[d];
  atomic_add_f32(&A1[(size_t)d * HDIM + f], w * H0[(size_t)s * HDIM + f]);
}

__global__ void relu_bias_kernel(float* __restrict__ A1, const float* __restrict__ b1, int n) {
  long long idx = (long long)blockIdx.x * 256 + threadIdx.x;
  if (idx >= (long long)n * HDIM) return;
  A1[idx] = fmaxf(A1[idx] + b1[idx & 63], 0.0f);
}

// ---------------------------------------------------------------------------
// GEMM2: H2[n x 7] = H1[n x 64] @ W2[64 x 7]  (N padded to 16)
// ---------------------------------------------------------------------------
__global__ __launch_bounds__(256) void gemm2_kernel(const float* __restrict__ H1,
                                                    const float* __restrict__ W2,
                                                    float* __restrict__ H2, int n) {
  __shared__ __bf16 sW[2 * 32 * 16];  // 2 K-steps, 1 tile, B-fragment packed (2 KB)
  int tid = threadIdx.x;
  for (int idx = tid; idx < HDIM * 16; idx += 256) {
    int k = idx >> 4, c = idx & 15;
    float v = (c < CDIM) ? W2[k * CDIM + c] : 0.0f;
    int s = k >> 5, jj = k & 31;
    int hk = (jj >> 3) & 1, j = (jj & 7) + ((jj >> 4) << 3);
    int lane = c + hk * 16;
    sW[((s * 32 + lane) * 16) + j] = (__bf16)v;
  }
  __syncthreads();

  int wave = tid >> 5, lane = tid & 31;
  int lane_m = lane & 15, half = lane >> 4;
  int row0 = blockIdx.x * 128 + wave * 16;

  int arow = row0 + lane_m;
  if (arow >= n) arow = n - 1;
  const float* hrow = H1 + (size_t)arow * HDIM;

  v8f acc{};
#pragma unroll
  for (int k0 = 0; k0 < HDIM; k0 += 32) {
    int kb = k0 + half * 8;
    v16bf a;
#pragma unroll
    for (int j = 0; j < 8; ++j) {
      a[j]     = (__bf16)hrow[kb + j];
      a[8 + j] = (__bf16)hrow[kb + 16 + j];
    }
    const v16bf b = *(const v16bf*)(sW + (size_t)((((k0 >> 5) * 32 + lane) * 16)));
    acc = __builtin_amdgcn_wmma_f32_16x16x32_bf16(
        false, a, false, b, (short)0, acc, false, false);
  }

#pragma unroll
  for (int i = 0; i < 8; ++i) {
    int r = row0 + half * 8 + i;
    if (r < n && lane_m < CDIM) H2[(size_t)r * CDIM + lane_m] = acc[i];
  }
}

// ---------------------------------------------------------------------------
// Aggregation, layer 2 (7 features) -> d_out
// ---------------------------------------------------------------------------
__global__ void agg2_init_kernel(const float* __restrict__ dinv, const float* __restrict__ H2,
                                 const float* __restrict__ b2, float* __restrict__ out, int n) {
  long long idx = (long long)blockIdx.x * 256 + threadIdx.x;
  if (idx >= (long long)n * CDIM) return;
  int i = (int)(idx / CDIM);
  int c = (int)(idx % CDIM);
  float di = dinv[i];
  out[idx] = di * di * H2[idx] + b2[c];
}

__global__ void agg2_edges_kernel(const long long* __restrict__ src,
                                  const long long* __restrict__ dst,
                                  const float* __restrict__ dinv,
                                  const float* __restrict__ H2,
                                  float* __restrict__ out, long long total) {
  long long tid = (long long)blockIdx.x * 256 + threadIdx.x;
  if (tid >= total) return;
  long long e = tid >> 3;
  int c = (int)(tid & 7);
  if (c >= CDIM) return;
  int s = (int)src[e], d = (int)dst[e];
  float w = dinv[s] * dinv[d];
  atomic_add_f32(&out[(size_t)d * CDIM + c], w * H2[(size_t)s * CDIM + c]);
}

// ---------------------------------------------------------------------------
extern "C" void kernel_launch(void* const* d_in, const int* in_sizes, int n_in,
                              void* d_out, int out_size, void* d_ws, size_t ws_size,
                              hipStream_t stream) {
  const float*     X  = (const float*)d_in[0];      // [n, 500]
  const float*     W1 = (const float*)d_in[1];      // [500, 64]
  const float*     b1 = (const float*)d_in[2];      // [64]
  const float*     W2 = (const float*)d_in[3];      // [64, 7]
  const float*     b2 = (const float*)d_in[4];      // [7]
  const long long* ei = (const long long*)d_in[5];  // [2, E] int64

  const int n = in_sizes[0] / FDIM;
  const int E = in_sizes[5] / 2;
  const long long* src = ei;
  const long long* dst = ei + E;

  // workspace layout (floats)
  float* ws   = (float*)d_ws;
  float* deg  = ws;                       // n
  float* dinv = deg + n;                  // n
  float* H0   = dinv + n;                 // n*64
  float* A1   = H0 + (size_t)n * HDIM;    // n*64
  float* H2   = A1 + (size_t)n * HDIM;    // n*7
  float* out  = (float*)d_out;            // n*7

  const int TB = 256;
  int gb_n    = (n + TB - 1) / TB;
  int gb_e    = (E + TB - 1) / TB;
  int gb_rows = (n + 127) / 128;
  long long t1  = (long long)E * 64;
  long long t2  = (long long)E * 8;
  long long nf1 = (long long)n * HDIM;
  long long nf2 = (long long)n * CDIM;
  int gb_t1  = (int)((t1 + TB - 1) / TB);
  int gb_t2  = (int)((t2 + TB - 1) / TB);
  int gb_nf1 = (int)((nf1 + TB - 1) / TB);
  int gb_nf2 = (int)((nf2 + TB - 1) / TB);

  // normalization
  init_deg_kernel<<<gb_n, TB, 0, stream>>>(deg, n);
  deg_edges_kernel<<<gb_e, TB, 0, stream>>>(dst, deg, E);
  dinv_kernel<<<gb_n, TB, 0, stream>>>(deg, dinv, n);

  // layer 1
  gemm1_kernel<<<gb_rows, TB, 0, stream>>>(X, W1, H0, n);
  agg1_init_kernel<<<gb_nf1, TB, 0, stream>>>(dinv, H0, A1, n);
  agg1_edges_kernel<<<gb_t1, TB, 0, stream>>>(src, dst, dinv, H0, A1, t1);
  relu_bias_kernel<<<gb_nf1, TB, 0, stream>>>(A1, b1, n);

  // layer 2
  gemm2_kernel<<<gb_rows, TB, 0, stream>>>(A1, W2, H2, n);
  agg2_init_kernel<<<gb_nf2, TB, 0, stream>>>(dinv, H2, b2, out, n);
  agg2_edges_kernel<<<gb_t2, TB, 0, stream>>>(src, dst, dinv, H2, out, t2);
}
